// Attention_Fusion_1_39092792328564
// MI455X (gfx1250) — matrix operand order repaired and checked by reading, not compiled, not run on hardware
//
#include <hip/hip_runtime.h>

typedef __attribute__((ext_vector_type(16))) _Float16 v16h;
typedef __attribute__((ext_vector_type(8)))  _Float16 v8h;
typedef __attribute__((ext_vector_type(2)))  _Float16 v2h;
typedef __attribute__((ext_vector_type(8)))  float    v8f;

#define DTOT 256
#define NH   8
#define DH   32
#define NLV  2
#define NPT  4
#define BB   4
#define HH   64
#define WW   64
#define HWSZ (HH * WW)          // 4096
#define LQ   (NLV * HWSZ)       // 8192
#define NTOK (BB * LQ)          // 32768

// ---------------------------------------------------------------------------
// Generic WMMA GEMM:  C[M x NT] = A[M x 256] * B[256 x NT] (+ bias)
//   AMODE 0: A gathered from transposed feature maps (input_v / input_i, f32)
//   AMODE 1: A read from f16 workspace, row-major [token][256]
//   SMODE 0: store f16 "value" tensor, layout [b][head][q][32]
//   SMODE 1: store f32 logits, row-major [token][NT]
//   SMODE 2: atomicAdd f32 into d_out[(b*256+n)*HW + (q%HW)]  (folds v+i sum)
// Block = 128 threads (4 waves); tile = 64(M) x 64(N); K-step = 32.
// LDS A is row-major [m][k]; LDS B is TRANSPOSED [n][k] so every wave32 WMMA
// fragment is exactly two aligned ds_load_b128's per operand.
// ---------------------------------------------------------------------------
template <int AMODE, int SMODE, int NT>
__global__ __launch_bounds__(128) void gemm_wmma(
    const float* __restrict__ Av, const float* __restrict__ Ai,
    const _Float16* __restrict__ Af16,
    const float* __restrict__ Bw, const float* __restrict__ bias,
    void* __restrict__ dst)
{
    __shared__ __align__(16) _Float16 Asm[64][40];  // [m][k], 80B pitch
    __shared__ __align__(16) _Float16 Bsm[64][40];  // [n][k], transposed

    const int t    = threadIdx.x;
    const int lane = t & 31;
    const int wid  = t >> 5;          // 0..3 -> 16-row strip of the 64-row tile
    const int g    = lane >> 4;       // half-wave group
    const int r    = lane & 15;

    const int tok0 = blockIdx.x * 64;
    const int n0   = blockIdx.y * 64;

    // A-source base (AMODE 0): tile of 64 tokens never crosses a batch/level
    // boundary (64 | 4096).
    const int bglob = tok0 / LQ;
    const int q0    = tok0 % LQ;
    const float* aptr = (q0 < HWSZ) ? Av : Ai;

    // Stagers: thread owns one m (or n) column and 16 consecutive k values.
    const int mld = t & 63;
    const int kb0 = (t >> 6) * 16;    // 0 or 16

    const float* apbase =
        (AMODE == 0)
            ? aptr + (size_t)bglob * DTOT * HWSZ + (size_t)kb0 * HWSZ +
                  (q0 % HWSZ) + mld
            : nullptr;
    const float* bpbase = Bw + (size_t)kb0 * NT + n0 + mld;

    v8f acc[4] = {};

    for (int k0 = 0; k0 < DTOT; k0 += 32) {
        // ---- stage A tile (64 tokens x 32 k) ----
        if (AMODE == 0) {
            const float* p = apbase + (size_t)k0 * HWSZ;
#pragma unroll
            for (int jj = 0; jj < 16; jj += 2) {
                const _Float16 h0 = (_Float16)p[(size_t)jj * HWSZ];
                const _Float16 h1 = (_Float16)p[(size_t)(jj + 1) * HWSZ];
                *(v2h*)&Asm[mld][kb0 + jj] = (v2h){h0, h1};
            }
        } else {
            const int m  = t >> 1;
            const int kh = (t & 1) * 16;
            const v8h* src =
                (const v8h*)&Af16[(size_t)(tok0 + m) * DTOT + k0 + kh];
            *(v8h*)&Asm[m][kh]     = src[0];
            *(v8h*)&Asm[m][kh + 8] = src[1];
        }
        // ---- stage B tile, transposed into [n][k] ----
        {
            const float* p = bpbase + (size_t)k0 * NT;
#pragma unroll
            for (int jj = 0; jj < 16; jj += 2) {
                const _Float16 h0 = (_Float16)p[(size_t)jj * NT];
                const _Float16 h1 = (_Float16)p[(size_t)(jj + 1) * NT];
                *(v2h*)&Bsm[mld][kb0 + jj] = (v2h){h0, h1};
            }
            if (k0 + 32 < DTOT)  // prefetch next weight slab
                __builtin_prefetch(p + (size_t)32 * NT, 0, 1);
        }
        __syncthreads();

        // ---- A fragment: k in [g*8,g*8+8) ++ [16+g*8,16+g*8+8) ----
        const int row = wid * 16 + r;
        const v8h alo = *(const v8h*)&Asm[row][g * 8];
        const v8h ahi = *(const v8h*)&Asm[row][g * 8 + 16];
        const v16h afrag = __builtin_shufflevector(
            alo, ahi, 0, 1, 2, 3, 4, 5, 6, 7, 8, 9, 10, 11, 12, 13, 14, 15);

        // ---- 4 N-tiles of 16: B fragment = k in [g*8,g*8+8) ++ [16+g*8,16+g*8+8) ----
#pragma unroll
        for (int tt = 0; tt < 4; ++tt) {
            const int col = tt * 16 + r;
            const v8h blo = *(const v8h*)&Bsm[col][g * 8];
            const v8h bhi = *(const v8h*)&Bsm[col][g * 8 + 16];
            const v16h bfrag = __builtin_shufflevector(
                blo, bhi, 0, 1, 2, 3, 4, 5, 6, 7, 8, 9, 10, 11, 12, 13, 14, 15);
            acc[tt] = __builtin_amdgcn_wmma_f32_16x16x32_f16(
                false, afrag, false, bfrag, (short)0, acc[tt], false, false);
        }
        __syncthreads();
    }

    // ---- store: C layout lane=(N=r, group g), VGPR j -> M = wid*16 + g*8 + j
#pragma unroll
    for (int tt = 0; tt < 4; ++tt) {
#pragma unroll
        for (int j = 0; j < 8; ++j) {
            const int m   = wid * 16 + g * 8 + j;
            const int n   = n0 + tt * 16 + r;
            const int tok = tok0 + m;
            const float val = acc[tt][j] + bias[n];
            if (SMODE == 0) {
                const int bg = tok / LQ, q = tok % LQ;
                const int head = n >> 5, d = n & 31;
                ((_Float16*)dst)[(((size_t)bg * NH + head) * LQ + q) * DH + d] =
                    (_Float16)val;
            } else if (SMODE == 1) {
                ((float*)dst)[(size_t)tok * NT + n] = val;
            } else {
                const int bg = tok / LQ, q = tok % LQ;
                atomicAdd(&((float*)dst)[((size_t)bg * DTOT + n) * HWSZ +
                                         (q % HWSZ)],
                          val);
            }
        }
    }
}

// ---------------------------------------------------------------------------
// Softmax over 8 (level,point) logits per head + bilinear sampling + weighted
// sum.  One wave per (token, head); lane = head channel d in [0,32).
// Logits/offsets are loaded once per wave and shared via lane shuffles.
// ---------------------------------------------------------------------------
__global__ __launch_bounds__(256) void sample_kernel(
    const float* __restrict__ off_ws,       // [token][128]
    const float* __restrict__ attn_ws,      // [token][64]
    const _Float16* __restrict__ value_ws,  // [b][head][q][32]
    _Float16* __restrict__ fused_ws)        // [token][256]
{
    const int wgid = blockIdx.x * 8 + (threadIdx.x >> 5);
    const int tok  = wgid >> 3;
    const int head = wgid & 7;
    const int d    = threadIdx.x & 31;

    const int bg  = tok / LQ;
    const int q   = tok % LQ;
    const int pos = q % HWSZ;
    const float refx = ((float)(pos % WW) + 0.5f) / (float)WW;
    const float refy = ((float)(pos / WW) + 0.5f) / (float)HH;

    // one load per lane, then broadcast via shuffles
    const float myLog = attn_ws[(size_t)tok * 64 + head * 8 + (d & 7)];
    const float myOff = off_ws[(size_t)tok * 128 + head * 16 + (d & 15)];

    float lg[8], mx = -1e30f;
#pragma unroll
    for (int j = 0; j < 8; ++j) {
        lg[j] = __shfl(myLog, j);
        mx = fmaxf(mx, lg[j]);
    }
    float sum = 0.f;
#pragma unroll
    for (int j = 0; j < 8; ++j) { lg[j] = __expf(lg[j] - mx); sum += lg[j]; }
    const float inv = 1.f / sum;

    float acc = 0.f;
#pragma unroll
    for (int l = 0; l < NLV; ++l) {
        const size_t vbase = ((size_t)bg * NH + head) * LQ + (size_t)l * HWSZ;
#pragma unroll
        for (int p = 0; p < NPT; ++p) {
            const int j = l * NPT + p;
            const float ox = __shfl(myOff, j * 2);
            const float oy = __shfl(myOff, j * 2 + 1);
            const float x = refx * WW + ox - 0.5f;
            const float y = refy * HH + oy - 0.5f;
            const float x0 = floorf(x), y0 = floorf(y);
            const float dx = x - x0, dy = y - y0;
            const int xi = (int)x0, yi = (int)y0;
            const float aw = lg[j] * inv;
            const int   cx[4] = {xi, xi + 1, xi,     xi + 1};
            const int   cy[4] = {yi, yi,     yi + 1, yi + 1};
            const float cw[4] = {(1 - dx) * (1 - dy), dx * (1 - dy),
                                 (1 - dx) * dy,       dx * dy};
#pragma unroll
            for (int c = 0; c < 4; ++c) {
                if (cx[c] >= 0 && cx[c] < WW && cy[c] >= 0 && cy[c] < HH) {
                    const float v =
                        (float)value_ws[(vbase + cy[c] * WW + cx[c]) * DH + d];
                    acc += aw * cw[c] * v;
                }
            }
        }
    }
    fused_ws[(size_t)tok * DTOT + head * DH + d] = (_Float16)acc;
}

// ---------------------------------------------------------------------------
extern "C" void kernel_launch(void* const* d_in, const int* in_sizes, int n_in,
                              void* d_out, int out_size, void* d_ws, size_t ws_size,
                              hipStream_t stream) {
    const float* inv   = (const float*)d_in[0];
    const float* ini   = (const float*)d_in[1];
    const float* Wv    = (const float*)d_in[2];
    const float* bv    = (const float*)d_in[3];
    const float* Woff  = (const float*)d_in[4];
    const float* boff  = (const float*)d_in[5];
    const float* Wattn = (const float*)d_in[6];
    const float* battn = (const float*)d_in[7];
    const float* Wout  = (const float*)d_in[8];
    const float* bout  = (const float*)d_in[9];

    char* ws = (char*)d_ws;
    _Float16* value_ws = (_Float16*)ws;                                  // 16 MiB
    float*    off_ws   = (float*)(ws + (size_t)16 * 1024 * 1024);        // 16 MiB
    float*    attn_ws  = (float*)(ws + (size_t)32 * 1024 * 1024);        //  8 MiB
    _Float16* fused_ws = (_Float16*)(ws + (size_t)40 * 1024 * 1024);     // 16 MiB

    hipMemsetAsync(d_out, 0, (size_t)out_size * sizeof(float), stream);

    const dim3 blk(128);
    // value / offset / attention projections off the transposed inputs
    gemm_wmma<0, 0, 256><<<dim3(NTOK / 64, 4), blk, 0, stream>>>(
        inv, ini, nullptr, Wv, bv, (void*)value_ws);
    gemm_wmma<0, 1, 128><<<dim3(NTOK / 64, 2), blk, 0, stream>>>(
        inv, ini, nullptr, Woff, boff, (void*)off_ws);
    gemm_wmma<0, 1, 64><<<dim3(NTOK / 64, 1), blk, 0, stream>>>(
        inv, ini, nullptr, Wattn, battn, (void*)attn_ws);

    // softmax + bilinear sampling + attention-weighted fuse
    sample_kernel<<<(NTOK * NH) / 8, 256, 0, stream>>>(
        off_ws, attn_ws, value_ws, fused_ws);

    // output projection; out_v + out_i folded in via atomicAdd
    gemm_wmma<1, 2, 256><<<dim3(NTOK / 64, 4), blk, 0, stream>>>(
        nullptr, nullptr, fused_ws, Wout, bout, d_out);
}
